// ViT_13606456393888
// MI455X (gfx1250) — matrix operand order repaired and checked by reading, not compiled, not run on hardware
//
#include <hip/hip_runtime.h>
#include <math.h>

#define DEV __device__ __forceinline__

typedef __attribute__((ext_vector_type(16))) __bf16 v16bf;
typedef __attribute__((ext_vector_type(8)))  __bf16 v8bf;
typedef __attribute__((ext_vector_type(8)))  float  v8f;

// 16-byte int vector + addrspace-qualified pointers matching the async builtin
// signature (param printed by hipcc as "int __vector(4) __device__ *").
typedef int v4i __attribute__((vector_size(16)));
typedef __attribute__((address_space(1))) v4i* as1_v4i_p;
typedef __attribute__((address_space(3))) v4i* as3_v4i_p;

constexpr int kB  = 16;
constexpr int kE  = 768;
constexpr int kH  = 12;
constexpr int kHD = 64;
constexpr int kN  = 197;   // tokens incl. cls
constexpr int kL  = 12;
constexpr int kFF = 3072;
constexpr int kMT = kB * kN;        // 3152 rows (transformer)
constexpr int kMP = kB * (kN - 1);  // 3136 rows (patches)

#define SQRT_E   27.712812921102035f
#define SQRT_HD  8.0f
#define ALPHA_COEF (0.05f * SQRT_E)

// ---------- bf16 helpers (raw u16 storage; RNE convert) ----------
DEV unsigned short f2bf(float x) {
  unsigned int u = __float_as_uint(x);
  unsigned int r = (u + 0x7fffu + ((u >> 16) & 1u)) >> 16;
  return (unsigned short)r;
}
DEV float bf2f(unsigned short h) { return __uint_as_float(((unsigned int)h) << 16); }

// intra-wave LDS RAW fence: DS ops are in-order per wave; asm pins compiler
// ordering and waits DScnt==0 (CDNA5 split counters).
DEV void lds_fence() { __asm__ volatile("s_wait_dscnt 0" ::: "memory"); }

// ---------- CDNA5 async global->LDS staging (ASYNCcnt path) ----------
#if __has_builtin(__builtin_amdgcn_global_load_async_to_lds_b128)
#define HAVE_ASYNC_LDS 1
#else
#define HAVE_ASYNC_LDS 0
#endif

DEV void stage_b128(const unsigned short* g, unsigned short* l) {
#if HAVE_ASYNC_LDS
  __builtin_amdgcn_global_load_async_to_lds_b128(
      (as1_v4i_p)(unsigned short*)g,   // strip const, reinterpret+addrspacecast to AS1 v4i*
      (as3_v4i_p)l,                    // LDS destination as AS3 v4i*
      0 /*imm offset*/, 0 /*cpol*/);
#else
  *(uint4*)l = *(const uint4*)g;       // sync fallback (still double-buffered)
#endif
}
DEV void async_wait1() {
#if __has_builtin(__builtin_amdgcn_s_wait_asynccnt)
  __builtin_amdgcn_s_wait_asynccnt(1);
#elif HAVE_ASYNC_LDS
  __asm__ volatile("s_wait_asynccnt 1" ::: "memory");
#endif
}
DEV void async_wait0() {
#if __has_builtin(__builtin_amdgcn_s_wait_asynccnt)
  __builtin_amdgcn_s_wait_asynccnt(0);
#elif HAVE_ASYNC_LDS
  __asm__ volatile("s_wait_asynccnt 0" ::: "memory");
#endif
}

// ---------- elementwise: f32 -> bf16 ----------
__global__ __launch_bounds__(256) void k_f2bf(const float* __restrict__ s,
                                              unsigned short* __restrict__ d, int n) {
  int i = blockIdx.x * 256 + threadIdx.x;
  if (i < n) d[i] = f2bf(s[i]);
}

// ---------- patch extraction -> bf16 [kMP, 768] ----------
__global__ __launch_bounds__(256) void k_patches(const float* __restrict__ x,
                                                 unsigned short* __restrict__ d) {
  int i = blockIdx.x * 256 + threadIdx.x;
  if (i >= kMP * kE) return;
  int e = i % kE;          // c*256 + py*16 + px   (3*16*16 == 768)
  int r = i / kE;
  int n = r % 196, b = r / 196;
  int c = e >> 8, rem = e & 255, py = rem >> 4, px = rem & 15;
  int gy = n / 14, gx = n % 14;
  size_t src = (((size_t)(b * 3 + c) * 224) + (size_t)(gy * 16 + py)) * 224 + (gx * 16 + px);
  d[i] = f2bf(x[src]);
}

// ---------- assemble t = [cls | patch_embed] + pos ----------
__global__ __launch_bounds__(256) void k_assemble(float* __restrict__ t,
                                                  const float* __restrict__ pe,
                                                  const float* __restrict__ cls,
                                                  const float* __restrict__ pos) {
  int i = blockIdx.x * 256 + threadIdx.x;
  if (i >= kMT * kE) return;
  int e = i % kE;
  int r = i / kE;
  int b = r / kN, nn = r % kN;
  if (nn == 0)
    t[i] = cls[e] + pos[e];
  else
    t[i] = pe[((size_t)(b * 196) + (nn - 1)) * kE + e] + pos[(size_t)nn * kE + e];
}

// ---------- WMMA GEMM: C[M,N] = A[M,K](bf16) * Bw[N,K](bf16)^T (+bias) ----------
// block = 256 thr = 8 waves; block tile 256(M) x 64(N); wave tile 32 x 64.
// B-tile (64x32 bf16 = 4KB) double-buffered in LDS via async global->LDS DMA.
// Ragged M edge handled by CLAMPING A-row indices (branch-free inner loop;
// WMMA executes unconditionally with EXEC all-ones); only the C store is guarded.
__global__ __launch_bounds__(256) void gemm_bf16nt(const unsigned short* __restrict__ A,
                                                   const unsigned short* __restrict__ Bw,
                                                   float* __restrict__ C,
                                                   const float* __restrict__ bias,
                                                   int M, int N, int K) {
  __shared__ unsigned short Bs[2][64 * 32];
  const int tid  = threadIdx.x;
  const int wave = tid >> 5, lane = tid & 31;
  const int ml   = lane & 15, sel = lane >> 4;
  const int nblk = blockIdx.x * 64;
  const int m0   = blockIdx.y * 256 + wave * 32;
  const bool act0 = (m0 < M);          // M % 16 == 0 for all shapes here
  const bool act1 = (m0 + 16 < M);
  const int row0 = (m0 + ml      < M) ? (m0 + ml)      : (M - 1);  // clamp
  const int row1 = (m0 + 16 + ml < M) ? (m0 + 16 + ml) : (M - 1);  // clamp

  v8f acc0[4], acc1[4];
#pragma unroll
  for (int j = 0; j < 4; ++j)
#pragma unroll
    for (int r = 0; r < 8; ++r) { acc0[j][r] = 0.f; acc1[j][r] = 0.f; }

  const unsigned short* Arow0 = A + (size_t)row0 * K;
  const unsigned short* Arow1 = A + (size_t)row1 * K;
  const int brow = tid >> 2;           // 0..63
  const int bseg = (tid & 3) * 8;      // 0,8,16,24
  const unsigned short* Bsrc = Bw + (size_t)(nblk + brow) * K + bseg;
  unsigned short* Ld[2] = { &Bs[0][brow * 32 + bseg], &Bs[1][brow * 32 + bseg] };

  const int nt = K / 32;
  stage_b128(Bsrc, Ld[0]);             // prologue: tile 0
  for (int i = 0; i < nt; ++i) {
    __syncthreads();                   // all waves done with buffer being overwritten
    if (i + 1 < nt) {
      stage_b128(Bsrc + (size_t)(i + 1) * 32, Ld[(i + 1) & 1]);
      async_wait1();                   // in-order: tile i has landed
    } else {
      async_wait0();
    }
    __syncthreads();                   // tile i visible to all waves
    const unsigned short* Bbuf = Bs[i & 1];
    const int k0 = i * 32;
    // A frag (16x32): elems 0..7 -> k0+sel*8+i ; 8..15 -> k0+16+sel*8+(i-8)
    const __bf16* ap0 = (const __bf16*)(Arow0 + k0 + sel * 8);
    v8bf a0lo = *(const v8bf*)ap0;
    v8bf a0hi = *(const v8bf*)(ap0 + 16);
    v16bf af0 = __builtin_shufflevector(a0lo, a0hi,
                  0, 1, 2, 3, 4, 5, 6, 7, 8, 9, 10, 11, 12, 13, 14, 15);
    const __bf16* ap1 = (const __bf16*)(Arow1 + k0 + sel * 8);
    v8bf a1lo = *(const v8bf*)ap1;
    v8bf a1hi = *(const v8bf*)(ap1 + 16);
    v16bf af1 = __builtin_shufflevector(a1lo, a1hi,
                  0, 1, 2, 3, 4, 5, 6, 7, 8, 9, 10, 11, 12, 13, 14, 15);
#pragma unroll
    for (int j = 0; j < 4; ++j) {
      // B frag (32x16): lane n = ml; elems 0..7 -> k = sel*8 + i ; 8..15 -> 16 + sel*8 + (i-8); reused by 2 WMMAs
      const __bf16* bp = (const __bf16*)(&Bbuf[(j * 16 + ml) * 32 + sel * 8]);
      v16bf bf = __builtin_shufflevector(*(const v8bf*)bp, *(const v8bf*)(bp + 16),
                  0, 1, 2, 3, 4, 5, 6, 7, 8, 9, 10, 11, 12, 13, 14, 15);
      acc0[j] = __builtin_amdgcn_wmma_f32_16x16x32_bf16(
          false, af0, false, bf, (short)0, acc0[j], false, false);
      acc1[j] = __builtin_amdgcn_wmma_f32_16x16x32_bf16(
          false, af1, false, bf, (short)0, acc1[j], false, false);
    }
  }

#pragma unroll
  for (int j = 0; j < 4; ++j) {
    int col = nblk + j * 16 + ml;
    float bb = bias ? bias[col] : 0.f;
    if (act0) {
#pragma unroll
      for (int r = 0; r < 8; ++r)   // C layout: VGPR r -> M=r (lanes<16) / M=8+r
        C[(size_t)(m0 + r + sel * 8) * N + col] = acc0[j][r] + bb;
    }
    if (act1) {
#pragma unroll
      for (int r = 0; r < 8; ++r)
        C[(size_t)(m0 + 16 + r + sel * 8) * N + col] = acc1[j][r] + bb;
    }
  }
}

// ---------- attention: one workgroup per (b,h) ----------
__global__ __launch_bounds__(256) void k_attention(const float* __restrict__ q,
                                                   const float* __restrict__ kk,
                                                   const float* __restrict__ vv,
                                                   const float* __restrict__ sqk,
                                                   float* __restrict__ out) {
  __shared__ unsigned short ks[kN * kHD];   // normalized*scaled K, bf16
  __shared__ unsigned short vs[kN * kHD];   // V, bf16
  __shared__ float qrow[8][kHD];
  __shared__ float probs[8][kN];
  const int bh = blockIdx.x;
  const int b = bh / kH, h = bh % kH;
  const int tid = threadIdx.x, wave = tid >> 5, lane = tid & 31;
  const int d0 = lane * 2;
  const float sc0 = sqk[h * kHD + d0] * SQRT_E;
  const float sc1 = sqk[h * kHD + d0 + 1] * SQRT_E;

  for (int n = wave; n < kN; n += 8) {
    size_t ro = ((size_t)(b * kN + n)) * kE + h * kHD;
    float k0 = kk[ro + d0], k1 = kk[ro + d0 + 1];
    float ss = k0 * k0 + k1 * k1;
#pragma unroll
    for (int m = 16; m >= 1; m >>= 1) ss += __shfl_xor(ss, m, 32);
    float inv = 1.0f / fmaxf(sqrtf(ss), 1e-6f);
    ks[n * kHD + d0]     = f2bf(k0 * inv * sc0);
    ks[n * kHD + d0 + 1] = f2bf(k1 * inv * sc1);
    vs[n * kHD + d0]     = f2bf(vv[ro + d0]);
    vs[n * kHD + d0 + 1] = f2bf(vv[ro + d0 + 1]);
  }
  __syncthreads();

  for (int n = wave; n < kN; n += 8) {
    size_t ro = ((size_t)(b * kN + n)) * kE + h * kHD;
    float q0 = q[ro + d0], q1 = q[ro + d0 + 1];
    float ss = q0 * q0 + q1 * q1;
#pragma unroll
    for (int m = 16; m >= 1; m >>= 1) ss += __shfl_xor(ss, m, 32);
    float inv = 1.0f / fmaxf(sqrtf(ss), 1e-6f);
    qrow[wave][d0]     = q0 * inv * sc0;
    qrow[wave][d0 + 1] = q1 * inv * sc1;
    lds_fence();
    for (int n2 = lane; n2 < kN; n2 += 32) {
      float acc = 0.f;
#pragma unroll 8
      for (int d = 0; d < kHD; ++d) acc += qrow[wave][d] * bf2f(ks[n2 * kHD + d]);
      probs[wave][n2] = acc * SQRT_HD;
    }
    lds_fence();
    float lmax = -1e30f;
    for (int n2 = 0; n2 < kN; ++n2) lmax = fmaxf(lmax, probs[wave][n2]);
    for (int n2 = lane; n2 < kN; n2 += 32)
      probs[wave][n2] = __expf(probs[wave][n2] - lmax);
    lds_fence();
    float lsum = 0.f;
    for (int n2 = 0; n2 < kN; ++n2) lsum += probs[wave][n2];
    float invs = 1.0f / lsum;
    float o0 = 0.f, o1 = 0.f;
    for (int n2 = 0; n2 < kN; ++n2) {
      float p = probs[wave][n2];
      o0 += p * bf2f(vs[n2 * kHD + d0]);
      o1 += p * bf2f(vs[n2 * kHD + d0 + 1]);
    }
    out[ro + d0]     = o0 * invs;
    out[ro + d0 + 1] = o1 * invs;
  }
}

// ---------- block reduction helper (256 threads, 8 waves) ----------
DEV float blockReduceSum(float v, float* sh) {
  __syncthreads();  // protect sh reuse across calls
#pragma unroll
  for (int m = 16; m >= 1; m >>= 1) v += __shfl_xor(v, m, 32);
  const int wave = threadIdx.x >> 5, lane = threadIdx.x & 31;
  if (lane == 0) sh[wave] = v;
  __syncthreads();
  float s = sh[0];
#pragma unroll
  for (int i = 1; i < 8; ++i) s += sh[i];
  return s;
}

// ---------- t = cosnorm(t + alpha*coef*(cosnorm(hx) - t)), per row ----------
__global__ __launch_bounds__(256) void k_resnorm(float* __restrict__ t,
                                                 const float* __restrict__ hx,
                                                 const float* __restrict__ alpha) {
  __shared__ float sh[8];
  const size_t row = blockIdx.x;
  const float* hr = hx + row * kE;
  float* tr = t + row * kE;
  float hv[3], tv[3], rv[3];
  float ss = 0.f;
#pragma unroll
  for (int i = 0; i < 3; ++i) {
    int c = threadIdx.x + i * 256;
    hv[i] = hr[c];
    tv[i] = tr[c];
    ss += hv[i] * hv[i];
  }
  float inv1 = 1.0f / fmaxf(sqrtf(blockReduceSum(ss, sh)), 1e-6f);
  float ss2 = 0.f;
#pragma unroll
  for (int i = 0; i < 3; ++i) {
    int c = threadIdx.x + i * 256;
    float a = alpha[c] * ALPHA_COEF;
    rv[i] = tv[i] + a * (hv[i] * inv1 - tv[i]);
    ss2 += rv[i] * rv[i];
  }
  float inv2 = 1.0f / fmaxf(sqrtf(blockReduceSum(ss2, sh)), 1e-6f);
#pragma unroll
  for (int i = 0; i < 3; ++i) tr[threadIdx.x + i * 256] = rv[i] * inv2;
}

// ---------- SwiGLU: bf16 out = u*s_u * silu(v*s_v*sqrtE) ----------
__global__ __launch_bounds__(256) void k_swiglu(const float* __restrict__ u,
                                                const float* __restrict__ v,
                                                const float* __restrict__ su,
                                                const float* __restrict__ sv,
                                                unsigned short* __restrict__ d) {
  int i = blockIdx.x * 256 + threadIdx.x;
  if (i >= kMT * kFF) return;
  int f = i % kFF;
  float uu = u[i] * su[f];
  float gg = v[i] * sv[f] * SQRT_E;
  float sig = 1.0f / (1.0f + __expf(-gg));
  d[i] = f2bf(uu * gg * sig);
}

// ---------- layer_norm(t)[:,1:] -> out[kMP, kE] ----------
__global__ __launch_bounds__(256) void k_layernorm(const float* __restrict__ t,
                                                   const float* __restrict__ w,
                                                   const float* __restrict__ bb,
                                                   float* __restrict__ out) {
  __shared__ float sh[8];
  int r = blockIdx.x;                 // 0..kMP-1
  int b = r / 196, n = r % 196;
  const float* x = t + ((size_t)(b * kN + 1 + n)) * kE;
  float* o = out + (size_t)r * kE;
  float v[3];
  float s = 0.f;
#pragma unroll
  for (int i = 0; i < 3; ++i) { v[i] = x[threadIdx.x + i * 256]; s += v[i]; }
  float mean = blockReduceSum(s, sh) * (1.0f / kE);
  float sq = 0.f;
#pragma unroll
  for (int i = 0; i < 3; ++i) { float d = v[i] - mean; sq += d * d; }
  float var = blockReduceSum(sq, sh) * (1.0f / kE);
  float inv = rsqrtf(var + 1e-6f);
#pragma unroll
  for (int i = 0; i < 3; ++i) {
    int c = threadIdx.x + i * 256;
    o[c] = (v[i] - mean) * inv * w[c] + bb[c];
  }
}

// ================= host orchestration =================
extern "C" void kernel_launch(void* const* d_in, const int* in_sizes, int n_in,
                              void* d_out, int out_size, void* d_ws, size_t ws_size,
                              hipStream_t stream) {
  (void)in_sizes; (void)n_in; (void)out_size; (void)ws_size;
  const float* x       = (const float*)d_in[0];
  const float* patch_w = (const float*)d_in[1];
  const float* patch_b = (const float*)d_in[2];
  const float* cls_tok = (const float*)d_in[3];
  const float* pos_emb = (const float*)d_in[4];
  const float* Wq      = (const float*)d_in[5];
  const float* Wk      = (const float*)d_in[6];
  const float* Wv      = (const float*)d_in[7];
  const float* Wo      = (const float*)d_in[8];
  const float* s_qk    = (const float*)d_in[9];
  const float* Wup     = (const float*)d_in[10];
  const float* Wgate   = (const float*)d_in[11];
  const float* Wdown   = (const float*)d_in[12];
  const float* s_u     = (const float*)d_in[13];
  const float* s_v     = (const float*)d_in[14];
  const float* alpha_A = (const float*)d_in[15];
  const float* alpha_M = (const float*)d_in[16];
  const float* fc_w    = (const float*)d_in[17];
  const float* fc_b    = (const float*)d_in[18];
  float* outp = (float*)d_out;

  char* base = (char*)d_ws;
  size_t off = 0;
  auto alloc = [&](size_t bytes) -> void* {
    size_t a = (off + 255) & ~(size_t)255;
    off = a + bytes;
    return base + a;
  };
  float* t    = (float*)alloc((size_t)kMT * kE * 4);
  float* qbuf = (float*)alloc((size_t)kMT * kE * 4);
  float* kbuf = (float*)alloc((size_t)kMT * kE * 4);
  float* vbuf = (float*)alloc((size_t)kMT * kE * 4);
  float* g0   = (float*)alloc((size_t)kMT * kFF * 4);
  float* g1   = (float*)alloc((size_t)kMT * kFF * 4);
  float* h1   = (float*)alloc((size_t)kMT * kE * 4);
  unsigned short* abf  = (unsigned short*)alloc((size_t)kMT * kFF * 2);
  unsigned short* wqb  = (unsigned short*)alloc((size_t)kE * kE * 2);
  unsigned short* wkb  = (unsigned short*)alloc((size_t)kE * kE * 2);
  unsigned short* wvb  = (unsigned short*)alloc((size_t)kE * kE * 2);
  unsigned short* wob  = (unsigned short*)alloc((size_t)kE * kE * 2);
  unsigned short* wupb = (unsigned short*)alloc((size_t)kFF * kE * 2);
  unsigned short* wgb  = (unsigned short*)alloc((size_t)kFF * kE * 2);
  unsigned short* wdb  = (unsigned short*)alloc((size_t)kE * kFF * 2);

  auto cvt = [&](const float* s, unsigned short* d, int n) {
    k_f2bf<<<(n + 255) / 256, 256, 0, stream>>>(s, d, n);
  };
  auto gemm = [&](const unsigned short* A, const unsigned short* Bm, float* C,
                  const float* bias, int M, int N, int K) {
    dim3 g(N / 64, (M + 255) / 256);
    gemm_bf16nt<<<g, 256, 0, stream>>>(A, Bm, C, bias, M, N, K);
  };

  // ---- patch embed ----
  cvt(patch_w, wqb, kE * 768);
  k_patches<<<(kMP * kE + 255) / 256, 256, 0, stream>>>(x, abf);
  gemm(abf, wqb, h1, patch_b, kMP, kE, 768);
  k_assemble<<<(kMT * kE + 255) / 256, 256, 0, stream>>>(t, h1, cls_tok, pos_emb);

  // ---- transformer layers ----
  for (int l = 0; l < kL; ++l) {
    cvt(Wq + (size_t)l * kE * kE, wqb, kE * kE);
    cvt(Wk + (size_t)l * kE * kE, wkb, kE * kE);
    cvt(Wv + (size_t)l * kE * kE, wvb, kE * kE);
    cvt(Wo + (size_t)l * kE * kE, wob, kE * kE);
    cvt(Wup + (size_t)l * kFF * kE, wupb, kFF * kE);
    cvt(Wgate + (size_t)l * kFF * kE, wgb, kFF * kE);
    cvt(Wdown + (size_t)l * kE * kFF, wdb, kE * kFF);

    // attention
    cvt(t, abf, kMT * kE);
    gemm(abf, wqb, qbuf, nullptr, kMT, kE, kE);
    gemm(abf, wkb, kbuf, nullptr, kMT, kE, kE);
    gemm(abf, wvb, vbuf, nullptr, kMT, kE, kE);
    k_attention<<<kB * kH, 256, 0, stream>>>(qbuf, kbuf, vbuf,
                                             s_qk + (size_t)l * kH * kHD, h1);
    cvt(h1, abf, kMT * kE);
    gemm(abf, wob, g0, nullptr, kMT, kE, kE);
    k_resnorm<<<kMT, 256, 0, stream>>>(t, g0, alpha_A + (size_t)l * kE);

    // MLP
    cvt(t, abf, kMT * kE);
    gemm(abf, wupb, g0, nullptr, kMT, kFF, kE);
    gemm(abf, wgb,  g1, nullptr, kMT, kFF, kE);
    k_swiglu<<<(kMT * kFF + 255) / 256, 256, 0, stream>>>(
        g0, g1, s_u + (size_t)l * kFF, s_v + (size_t)l * kFF, abf);
    gemm(abf, wdb, h1, nullptr, kMT, kE, kFF);
    k_resnorm<<<kMT, 256, 0, stream>>>(t, h1, alpha_M + (size_t)l * kE);

    if (l == kL - 2)
      k_layernorm<<<kMP, 256, 0, stream>>>(t, fc_w, fc_b, outp);                      // nodes
    if (l == kL - 1)
      k_layernorm<<<kMP, 256, 0, stream>>>(t, fc_w, fc_b, outp + (size_t)kMP * kE);   // edges
  }
}
